// PredCells_33088428048700
// MI455X (gfx1250) — matrix operand order repaired and checked by reading, not compiled, not run on hardware
//
#include <hip/hip_runtime.h>
#include <math.h>

// ---------------------------------------------------------------------------
// PredCells scan for MI455X (gfx1250): bf16 WMMA GEMVs + persistent kernel
// with grid barriers. Weight-bandwidth-bound problem; weights packed to bf16
// once per launch (they live in the 192MB L2 afterwards).
// ---------------------------------------------------------------------------

#define HDIM 1024
#define CDIM 56
#define TSTEPS 64

typedef __bf16 bf16;
typedef bf16  v16bf __attribute__((ext_vector_type(16)));
typedef bf16  v8bf  __attribute__((ext_vector_type(8)));
typedef float v8f   __attribute__((ext_vector_type(8)));

__device__ __forceinline__ bf16 tobf(float x) {
  unsigned u = __float_as_uint(x);
  unsigned r = u + 0x7FFFu + ((u >> 16) & 1u);   // round-to-nearest-even
  unsigned short h = (unsigned short)(r >> 16);
  bf16 b;
  __builtin_memcpy(&b, &h, sizeof(h));
  return b;
}

__device__ __forceinline__ float sigm(float x) { return 1.0f / (1.0f + expf(-x)); }

// ---------------------------------------------------------------------------
// Grid-wide barrier (sense/generation, agent scope). 32 WGs -> co-resident.
// ---------------------------------------------------------------------------
__device__ __forceinline__ void grid_barrier(unsigned* cnt, unsigned* gen) {
  __syncthreads();
  if (threadIdx.x == 0) {
    __threadfence();
    unsigned g = __hip_atomic_load(gen, __ATOMIC_RELAXED, __HIP_MEMORY_SCOPE_AGENT);
    unsigned prev = __hip_atomic_fetch_add(cnt, 1u, __ATOMIC_ACQ_REL, __HIP_MEMORY_SCOPE_AGENT);
    if (prev + 1u == gridDim.x) {
      __hip_atomic_store(cnt, 0u, __ATOMIC_RELAXED, __HIP_MEMORY_SCOPE_AGENT);
      __hip_atomic_fetch_add(gen, 1u, __ATOMIC_ACQ_REL, __HIP_MEMORY_SCOPE_AGENT);
    } else {
      while (__hip_atomic_load(gen, __ATOMIC_ACQUIRE, __HIP_MEMORY_SCOPE_AGENT) == g) {
        __builtin_amdgcn_s_sleep(2);
      }
    }
    __threadfence();
  }
  __syncthreads();
}

// ---------------------------------------------------------------------------
// GEMV tile via v_wmma_f32_16x16x32_bf16.
// A (16x32): every row m = x[k0..k0+31] (broadcast)  -> every C row equals y.
// B (32x16): B[k][n] = W[n0+n][k]; lane holds col n=lane&15, K=(lane>>4)*16..+15
//            -> 16 contiguous bf16 from the row-major weight row (32B load).
// C (16x16 f32 accum): c[0] on lanes 0..15 = y[n0+lane].
// ---------------------------------------------------------------------------
__device__ __forceinline__ float gemv_tile16(const bf16* __restrict__ W, int ldw,
                                             const bf16* __restrict__ x, int K, int n0) {
  const unsigned lane = threadIdx.x & 31u;
  const unsigned hsel = lane >> 4;          // 0: lanes 0-15, 1: lanes 16-31
  const unsigned aoff = hsel * 16u;         // A-operand K offset for this half
  const bf16* wrow = W + (size_t)(n0 + (int)(lane & 15u)) * (size_t)ldw + hsel * 16u;
  v8f c = {};
  for (int k0 = 0; k0 < K; k0 += 32) {
    v8bf a0 = *(const v8bf*)(x + k0 + aoff);        // K = aoff .. aoff+7
    v8bf a1 = *(const v8bf*)(x + k0 + aoff + 8);    // K = aoff+8 .. aoff+15
    v16bf a = __builtin_shufflevector(a0, a1, 0, 1, 2, 3, 4, 5, 6, 7,
                                      8, 9, 10, 11, 12, 13, 14, 15);
    v16bf b = *(const v16bf*)(wrow + k0);
    c = __builtin_amdgcn_wmma_f32_16x16x32_bf16(false, a, false, b,
                                                (short)0, c, false, false);
  }
  return c[0];
}

// ---------------------------------------------------------------------------
struct Params {
  // fp32 inputs (device pointers from harness)
  const float* x;                     // [T, C]
  const float* W0w; const float* W0b; // [H,C],[H]
  const float* W1b; const float* W2b; // [H]
  const float* b1;  const float* b2;  const float* b3;   // [4H]
  const float* V1b; const float* V2b; const float* V3b;  // [C]/[H]/[H]
  const int* iternum;
  // packed bf16 weights (workspace)
  const bf16 *V1p;                    // [64, H]  (N padded 56->64, zero rows)
  const bf16 *W1p, *W2p, *V2p, *V3p;  // [H, H]
  const bf16 *Wih1p, *Wih2p;          // [4H, 2H]
  const bf16 *Wih3p;                  // [4H, H]
  // fp32 state (workspace)
  float* zbuf;                        // [4H] pre-activation scratch
  float *recon1, *recon2, *recon3;    // [64],[H],[H]
  // bf16 state (workspace)
  bf16 *in1b, *in2b;                  // [2H]: [BU, TD]
  bf16 *in3b;                         // [H] : BU2
  bf16 *s1b, *s2b, *s3b;              // [H]
  bf16 *ntd1b, *ntd2b;                // [H]
  // sync / output
  unsigned* cnt; unsigned* gen; float* loss;
  float* out;
};

__global__ __launch_bounds__(256) void predcells_pack_bf16(
    const float* __restrict__ src, bf16* __restrict__ dst,
    int N, int K, int Npad, int Kpad) {
  long long total = (long long)Npad * (long long)Kpad;
  long long stride = (long long)gridDim.x * blockDim.x;
  for (long long i = (long long)blockIdx.x * blockDim.x + threadIdx.x;
       i < total; i += stride) {
    int n = (int)(i / Kpad);
    int k = (int)(i % Kpad);
    float v = (n < N && k < K) ? src[(long long)n * K + k] : 0.0f;
    dst[i] = tobf(v);
  }
}

__global__ __launch_bounds__(256) void predcells_init_state(
    unsigned* hdr, float* recon1, float* recon2, float* recon3,
    bf16* in1b, bf16* in2b) {
  int i = blockIdx.x * blockDim.x + threadIdx.x;
  if (i < 64) { hdr[i] = 0u; recon1[i] = 0.0f; }
  if (i < HDIM) {
    recon2[i] = 0.0f;
    recon3[i] = 0.0f;
    in1b[HDIM + i] = tobf(0.0f);   // TD1 = 0
    in2b[HDIM + i] = tobf(0.0f);   // TD2 = 0
  }
}

// ---------------------------------------------------------------------------
// Persistent scan kernel: 32 blocks x 256 threads = 256 wave32s
// (exactly one wave per 16-row tile of the 4096-row LSTM GEMVs).
// ---------------------------------------------------------------------------
__global__ __launch_bounds__(256) void predcells_scan(Params p) {
  const int H = HDIM, C = CDIM;
  const unsigned lane = threadIdx.x & 31u;
  const int wave = (int)((blockIdx.x * blockDim.x + threadIdx.x) >> 5);
  const int nwaves = (int)((gridDim.x * blockDim.x) >> 5);
  const int gtid = blockIdx.x * blockDim.x + threadIdx.x;
  const int nthr = gridDim.x * blockDim.x;
  const float lam = (*p.iternum <= 1000) ? 1e-4f : 1e-2f;
  const float lam2 = lam * lam;
  __shared__ float td0[64];

  for (int t = 0; t < TSTEPS; ++t) {
    // ---- Stage A (block 0 only): nTD0 = x_t - recon1; loss += |nTD0|;
    //      BU0 = W0 @ nTD0 + W0_b -> in1b[0:H] (scalar fp32, K=56 is tiny)
    if (blockIdx.x == 0) {
      if (threadIdx.x < 64) {
        float v = 0.0f;
        if ((int)threadIdx.x < C) {
          v = p.x[t * C + threadIdx.x] - p.recon1[threadIdx.x];
          unsafeAtomicAdd(p.loss, fabsf(v));
        }
        td0[threadIdx.x] = v;
      }
      __syncthreads();
      for (int n = threadIdx.x; n < H; n += blockDim.x) {
        const float* wr = p.W0w + (size_t)n * C;
        float acc = p.W0b[n];
#pragma unroll 8
        for (int j = 0; j < C; ++j) acc += wr[j] * td0[j];
        p.in1b[n] = tobf(acc);
      }
    }
    grid_barrier(p.cnt, p.gen);

    // ---- Stage B1: z1 = Wih1 @ [BU0, TD1]   (4096x2048, 256 tiles)
    for (int tile = wave; tile < 4 * H / 16; tile += nwaves) {
      float y = gemv_tile16(p.Wih1p, 2 * H, p.in1b, 2 * H, tile * 16);
      if (lane < 16) p.zbuf[tile * 16 + (int)lane] = y;
    }
    grid_barrier(p.cnt, p.gen);

    // ---- Stage B2: s1 = act(z1 + b1); nTD1 = s1 - recon2; loss += lam|nTD1|
    if (gtid < H) {
      int h = gtid;
      float zi = p.zbuf[h] + p.b1[h];
      float zg = p.zbuf[2 * H + h] + p.b1[2 * H + h];
      float zo = p.zbuf[3 * H + h] + p.b1[3 * H + h];
      float s = sigm(zo) * tanhf(sigm(zi) * tanhf(zg));
      float ntd = s - p.recon2[h];
      p.s1b[h] = tobf(s);
      p.ntd1b[h] = tobf(ntd);
      unsafeAtomicAdd(p.loss, lam * fabsf(ntd));
    }
    grid_barrier(p.cnt, p.gen);

    // ---- Stage C: BU1 = W1 @ nTD1 + W1_b -> in2b[0:H];
    //      recon1 = V1 @ s1 + V1_b; refresh TD1 half of in1b
    for (int i = gtid; i < H; i += nthr) p.in1b[H + i] = p.ntd1b[i];
    for (int tile = wave; tile < H / 16 + 4; tile += nwaves) {
      if (tile < H / 16) {
        float y = gemv_tile16(p.W1p, H, p.ntd1b, H, tile * 16);
        int n = tile * 16 + (int)lane;
        if (lane < 16) p.in2b[n] = tobf(y + p.W1b[n]);
      } else {
        int t2 = tile - H / 16;
        float y = gemv_tile16(p.V1p, H, p.s1b, H, t2 * 16);
        int n = t2 * 16 + (int)lane;
        if (lane < 16 && n < C) p.recon1[n] = y + p.V1b[n];
      }
    }
    grid_barrier(p.cnt, p.gen);

    // ---- Stage D1: z2 = Wih2 @ [BU1, TD2]
    for (int tile = wave; tile < 4 * H / 16; tile += nwaves) {
      float y = gemv_tile16(p.Wih2p, 2 * H, p.in2b, 2 * H, tile * 16);
      if (lane < 16) p.zbuf[tile * 16 + (int)lane] = y;
    }
    grid_barrier(p.cnt, p.gen);

    // ---- Stage D2: s2 = act(z2 + b2); nTD2 = s2 - recon3; loss += lam^2|nTD2|
    if (gtid < H) {
      int h = gtid;
      float zi = p.zbuf[h] + p.b2[h];
      float zg = p.zbuf[2 * H + h] + p.b2[2 * H + h];
      float zo = p.zbuf[3 * H + h] + p.b2[3 * H + h];
      float s = sigm(zo) * tanhf(sigm(zi) * tanhf(zg));
      float ntd = s - p.recon3[h];
      p.s2b[h] = tobf(s);
      p.ntd2b[h] = tobf(ntd);
      unsafeAtomicAdd(p.loss, lam2 * fabsf(ntd));
    }
    grid_barrier(p.cnt, p.gen);

    // ---- Stage E: BU2 = W2 @ nTD2 + W2_b -> in3b;
    //      recon2 = V2 @ s2 + V2_b; refresh TD2 half of in2b
    for (int i = gtid; i < H; i += nthr) p.in2b[H + i] = p.ntd2b[i];
    for (int tile = wave; tile < 2 * (H / 16); tile += nwaves) {
      if (tile < H / 16) {
        float y = gemv_tile16(p.W2p, H, p.ntd2b, H, tile * 16);
        int n = tile * 16 + (int)lane;
        if (lane < 16) p.in3b[n] = tobf(y + p.W2b[n]);
      } else {
        int t2 = tile - H / 16;
        float y = gemv_tile16(p.V2p, H, p.s2b, H, t2 * 16);
        int n = t2 * 16 + (int)lane;
        if (lane < 16) p.recon2[n] = y + p.V2b[n];
      }
    }
    grid_barrier(p.cnt, p.gen);

    // ---- Stage F1: z3 = Wih3 @ BU2    (4096x1024)
    for (int tile = wave; tile < 4 * H / 16; tile += nwaves) {
      float y = gemv_tile16(p.Wih3p, H, p.in3b, H, tile * 16);
      if (lane < 16) p.zbuf[tile * 16 + (int)lane] = y;
    }
    grid_barrier(p.cnt, p.gen);

    // ---- Stage F2: s3 = act(z3 + b3)
    if (gtid < H) {
      int h = gtid;
      float zi = p.zbuf[h] + p.b3[h];
      float zg = p.zbuf[2 * H + h] + p.b3[2 * H + h];
      float zo = p.zbuf[3 * H + h] + p.b3[3 * H + h];
      float s = sigm(zo) * tanhf(sigm(zi) * tanhf(zg));
      p.s3b[h] = tobf(s);
    }
    grid_barrier(p.cnt, p.gen);

    // ---- Stage G: recon3 = V3 @ s3 + V3_b
    for (int tile = wave; tile < H / 16; tile += nwaves) {
      float y = gemv_tile16(p.V3p, H, p.s3b, H, tile * 16);
      int n = tile * 16 + (int)lane;
      if (lane < 16) p.recon3[n] = y + p.V3b[n];
    }
    grid_barrier(p.cnt, p.gen);
  }

  if (gtid == 0) p.out[0] = *p.loss;
}

// ---------------------------------------------------------------------------
extern "C" void kernel_launch(void* const* d_in, const int* in_sizes, int n_in,
                              void* d_out, int out_size, void* d_ws, size_t ws_size,
                              hipStream_t stream) {
  (void)in_sizes; (void)n_in; (void)out_size; (void)ws_size;
  const int H = HDIM, C = CDIM;

  char* ws = (char*)d_ws;
  size_t cur = 0;
  auto take = [&](size_t bytes) -> void* {
    size_t o = (cur + 255) & ~(size_t)255;
    cur = o + bytes;
    return (void*)(ws + o);
  };

  unsigned* hdr   = (unsigned*)take(256);          // [0]=cnt [1]=gen [2]=loss
  float* zbuf     = (float*)take((size_t)4 * H * 4);
  float* recon1   = (float*)take(64 * 4);
  float* recon2   = (float*)take((size_t)H * 4);
  float* recon3   = (float*)take((size_t)H * 4);
  bf16* in1b      = (bf16*)take((size_t)2 * H * 2);
  bf16* in2b      = (bf16*)take((size_t)2 * H * 2);
  bf16* in3b      = (bf16*)take((size_t)H * 2);
  bf16* s1b       = (bf16*)take((size_t)H * 2);
  bf16* s2b       = (bf16*)take((size_t)H * 2);
  bf16* s3b       = (bf16*)take((size_t)H * 2);
  bf16* ntd1b     = (bf16*)take((size_t)H * 2);
  bf16* ntd2b     = (bf16*)take((size_t)H * 2);
  bf16* V1p       = (bf16*)take((size_t)64 * H * 2);
  bf16* W1p       = (bf16*)take((size_t)H * H * 2);
  bf16* W2p       = (bf16*)take((size_t)H * H * 2);
  bf16* V2p       = (bf16*)take((size_t)H * H * 2);
  bf16* V3p       = (bf16*)take((size_t)H * H * 2);
  bf16* Wih1p     = (bf16*)take((size_t)4 * H * 2 * H * 2);   // 16 MB
  bf16* Wih2p     = (bf16*)take((size_t)4 * H * 2 * H * 2);   // 16 MB
  bf16* Wih3p     = (bf16*)take((size_t)4 * H * H * 2);       // 8 MB

  // ---- pack weights fp32 -> bf16 (row-major kept; WMMA B-tile loads are
  //      16 contiguous bf16 per lane, so no layout swizzle needed)
  dim3 pb(256), pg(2048);
  predcells_pack_bf16<<<pg, pb, 0, stream>>>((const float*)d_in[13], V1p, C, H, 64, H);
  predcells_pack_bf16<<<pg, pb, 0, stream>>>((const float*)d_in[3],  W1p, H, H, H, H);
  predcells_pack_bf16<<<pg, pb, 0, stream>>>((const float*)d_in[5],  W2p, H, H, H, H);
  predcells_pack_bf16<<<pg, pb, 0, stream>>>((const float*)d_in[15], V2p, H, H, H, H);
  predcells_pack_bf16<<<pg, pb, 0, stream>>>((const float*)d_in[17], V3p, H, H, H, H);
  predcells_pack_bf16<<<pg, pb, 0, stream>>>((const float*)d_in[7],  Wih1p, 4 * H, 2 * H, 4 * H, 2 * H);
  predcells_pack_bf16<<<pg, pb, 0, stream>>>((const float*)d_in[9],  Wih2p, 4 * H, 2 * H, 4 * H, 2 * H);
  predcells_pack_bf16<<<pg, pb, 0, stream>>>((const float*)d_in[11], Wih3p, 4 * H, H, 4 * H, H);

  predcells_init_state<<<4, 256, 0, stream>>>(hdr, recon1, recon2, recon3, in1b, in2b);

  Params p;
  p.x    = (const float*)d_in[0];
  p.W0w  = (const float*)d_in[1];  p.W0b = (const float*)d_in[2];
  p.W1b  = (const float*)d_in[4];  p.W2b = (const float*)d_in[6];
  p.b1   = (const float*)d_in[8];  p.b2  = (const float*)d_in[10]; p.b3 = (const float*)d_in[12];
  p.V1b  = (const float*)d_in[14]; p.V2b = (const float*)d_in[16]; p.V3b = (const float*)d_in[18];
  p.iternum = (const int*)d_in[19];
  p.V1p = V1p; p.W1p = W1p; p.W2p = W2p; p.V2p = V2p; p.V3p = V3p;
  p.Wih1p = Wih1p; p.Wih2p = Wih2p; p.Wih3p = Wih3p;
  p.zbuf = zbuf; p.recon1 = recon1; p.recon2 = recon2; p.recon3 = recon3;
  p.in1b = in1b; p.in2b = in2b; p.in3b = in3b;
  p.s1b = s1b; p.s2b = s2b; p.s3b = s3b;
  p.ntd1b = ntd1b; p.ntd2b = ntd2b;
  p.cnt = hdr + 0; p.gen = hdr + 1; p.loss = (float*)(hdr + 2);
  p.out = (float*)d_out;

  // 32 blocks x 256 threads = 256 wave32s: one wave per 16-row tile of the
  // 4096-row LSTM GEMVs; grid barriers sequence the dependent stages.
  predcells_scan<<<32, 256, 0, stream>>>(p);
}
